// BBN_Layer_11897059410343
// MI455X (gfx1250) — matrix operand order repaired and checked by reading, not compiled, not run on hardware
//
#include <hip/hip_runtime.h>
#include <hip/hip_bf16.h>
#include <stdint.h>

typedef _Float16 half_t;
typedef __attribute__((ext_vector_type(16))) _Float16 v16h;
typedef __attribute__((ext_vector_type(8)))  float    v8f;

// Problem constants (from reference setup)
#define BATCH   8
#define HW      256
#define PIX     (HW*HW)          // 65536
#define OUT_HW  225
#define NPARTS  64
#define FS      32
#define NPLANES 15               // 0..3 -> numerator, 4..14 -> denominator
#define NSTEPS  (NPLANES*FS)     // 480 K-steps of K=32 each

// ---------------------------------------------------------------------------
// Kernel 1: build 15 derived f16 planes per batch image.
//   P0..2  = image*ga            (filter pf_c)        [numerator]
//   P3     = -sum_c image*ga*bg  (filter pa)          [numerator]
//   P4..6  = ga^2                (filter pf_c^2)
//   P7..9  = 2*aA*ga             (filter pf_c)
//   P10..12= 2*ga^2*bg           (filter pf_c*(1-pa))
//   P13    = sum_c (ga*bg)^2     (filter pa^2)
//   P14    = -sum_c (2(ga*bg)^2 + 2*aA*ga*bg)  (filter pa)
// ---------------------------------------------------------------------------
__global__ void prep_planes(const float* __restrict__ image,
                            const float* __restrict__ fa,
                            const float* __restrict__ aA,
                            const float* __restrict__ bg,
                            half_t* __restrict__ planes) {
    int tid = blockIdx.x * 256 + threadIdx.x;     // 0 .. 8*65536-1
    int b   = tid >> 16;
    int pix = tid & 65535;
    size_t ib = (size_t)b * (3 * PIX) + pix;
    half_t* pb = planes + (size_t)b * NPLANES * PIX + pix;
    float P3 = 0.f, P13 = 0.f, P14 = 0.f;
#pragma unroll
    for (int c = 0; c < 3; ++c) {
        float im = image[ib + (size_t)c * PIX];
        float f  = fa[ib + (size_t)c * PIX];
        float a  = aA[ib + (size_t)c * PIX];
        float g  = bg[ib + (size_t)c * PIX];
        float ga = 1.f - f;
        float iga = im * ga;
        float gb  = ga * g;
        pb[(size_t)(0  + c) * PIX] = (half_t)iga;
        pb[(size_t)(4  + c) * PIX] = (half_t)(ga * ga);
        pb[(size_t)(7  + c) * PIX] = (half_t)(2.f * a * ga);
        pb[(size_t)(10 + c) * PIX] = (half_t)(2.f * ga * ga * g);
        P3  -= iga * g;
        P13 += gb * gb;
        P14 -= 2.f * gb * gb + 2.f * a * gb;
    }
    pb[(size_t)3  * PIX] = (half_t)P3;
    pb[(size_t)13 * PIX] = (half_t)P13;
    pb[(size_t)14 * PIX] = (half_t)P14;
}

// ---------------------------------------------------------------------------
// Kernel 2: build filter bank pre-swizzled into B-fragment layout.
// B (32x16 f16, K x N): lanes 0-15 hold K=0..7,16..23 (N=lane), lanes 16-31 hold
// K=8..15,24..31 (N=lane-16); half s is K = 8*(lane>>4) + (s&7) + 16*(s>>3).
// Fragment (step = q*32+i, n-tile t): 32 lanes x 16 halfs contiguous (1 KB).
// ---------------------------------------------------------------------------
__global__ void prep_weights(const float* __restrict__ parts,
                             half_t* __restrict__ wbuf) {
    int tid  = blockIdx.x * 256 + threadIdx.x;    // 0 .. 983039
    int slot = tid & 15;
    int lane = (tid >> 4) & 31;
    int t    = (tid >> 9) & 3;
    int i    = (tid >> 11) & 31;
    int q    = tid >> 16;                          // 0..14
    int K = (slot & 7) | ((lane >> 4) << 3) | ((slot >> 3) << 4); // 0..31  (j index)
    int N = lane & 15;
    int p = t * 16 + N;                            // part index
    const float* pp = parts + (size_t)p * 4096;    // (64,4,32,32)
    int ij = i * 32 + K;
    float pa = pp[3 * 1024 + ij];
    float w;
    if (q < 3)        w = pp[q * 1024 + ij] * pa;                         // pf
    else if (q == 3)  w = pa;
    else if (q < 7)   { float f = pp[(q - 4) * 1024 + ij] * pa; w = f * f; } // pf^2
    else if (q < 10)  w = pp[(q - 7) * 1024 + ij] * pa;                   // pf
    else if (q < 13)  { float f = pp[(q - 10) * 1024 + ij] * pa; w = f * (1.f - pa); }
    else if (q == 13) w = pa * pa;
    else              w = pa;
    wbuf[((size_t)(q * 32 + i) * 4 + t) * 512 + lane * 16 + slot] = (half_t)w;
}

// ---------------------------------------------------------------------------
// Kernel 3: per-batch scalar reductions.
//   scal[b*4+0] = sum(image*aA) + sum(image*ga*bg)        (numerator offset)
//   scal[b*4+1] = sum(aA^2) + sum((ga*bg)^2) + sum(2*aA*ga*bg)
//   scal[b*4+2] = sum(image^2)                            (I_norm)
// ---------------------------------------------------------------------------
__global__ void reduce_scalars(const float* __restrict__ image,
                               const float* __restrict__ fa,
                               const float* __restrict__ aA,
                               const float* __restrict__ bg,
                               float* __restrict__ scal) {
    int b = blockIdx.x;
    size_t base = (size_t)b * (3 * PIX);
    float s0 = 0.f, s1 = 0.f, s2 = 0.f;
    for (int idx = threadIdx.x; idx < 3 * PIX; idx += 256) {
        float im = image[base + idx];
        float f  = fa[base + idx];
        float a  = aA[base + idx];
        float g  = bg[base + idx];
        float ga = 1.f - f;
        float gb = ga * g;
        s0 += im * a + im * gb;
        s1 += a * a + gb * gb + 2.f * a * gb;
        s2 += im * im;
    }
    __shared__ float r0[256], r1[256], r2[256];
    int tx = threadIdx.x;
    r0[tx] = s0; r1[tx] = s1; r2[tx] = s2;
    __syncthreads();
    for (int off = 128; off > 0; off >>= 1) {
        if (tx < off) { r0[tx] += r0[tx + off]; r1[tx] += r1[tx + off]; r2[tx] += r2[tx + off]; }
        __syncthreads();
    }
    if (tx == 0) { scal[b * 4 + 0] = r0[0]; scal[b * 4 + 1] = r1[0]; scal[b * 4 + 2] = r2[0]; }
}

// ---------------------------------------------------------------------------
// Kernel 4: implicit-GEMM conv via v_wmma_f32_16x16x32_f16.
// Block = (batch b, output row y0, 128 output x-pixels) x all 64 parts.
// 8 waves: wave w -> x sub-tile w (16 pixels); each wave computes ALL 4
// part-tiles -> 4 WMMA per wave per K-step (32 per block per step).
// Per K-step (plane q, kernel row i):
//   * 160-half pixel strip staged into LDS twice (raw + shifted-by-1) so all
//     lanes' A-fragment dword loads are 4B-aligned.
//   * 4KB weight block staged into LDS cooperatively (1 b128 load+store per
//     thread); waves read B fragments as aligned ds_load_b128 pairs.
//   * Double-buffered LDS -> exactly one barrier per K-step.
// ---------------------------------------------------------------------------
__global__ __launch_bounds__(256) void conv_main(
        const half_t* __restrict__ planes,
        const half_t* __restrict__ wbuf,
        const float*  __restrict__ scal,
        float* __restrict__ out) {
    __shared__ __align__(16) half_t strip[2][2][168];   // [buf][raw/shifted][..160]
    __shared__ __align__(16) half_t wlds[2][2048];      // [buf][4 tiles * 512]

    const int tid  = threadIdx.x;
    const int lane = tid & 31;
    const int wv   = tid >> 5;          // x sub-tile (0..7)
    const int b    = blockIdx.z;
    const int y0   = blockIdx.y;
    const int x0   = blockIdx.x * 128;

    // A-fragment addressing (16-bit A 16x32 layout):
    // lane<16: M=lane,    K in {0..7, 16..23}; lane>=16: M=lane-16, K in {8..15, 24..31}
    const int Ml   = lane & 15;
    const int hiL  = lane >> 4;
    const int sel  = Ml & 1;                    // odd M rows read the shifted copy
    const int abase = wv * 16 + Ml + hiL * 8 - sel;

    v8f accN[4] = {{}, {}, {}, {}};
    v8f accD[4] = {{}, {}, {}, {}};
    const uint4* wg = (const uint4*)wbuf;       // 256 x 16B per K-step
    const half_t* pbase = planes + (size_t)(b * NPLANES) * PIX;

    for (int q = 0; q < NPLANES; ++q) {
        const half_t* prow = pbase + (size_t)q * PIX + (size_t)y0 * HW;
        for (int i = 0; i < FS; ++i) {
            const int step = q * 32 + i;
            const int buf  = step & 1;

            // ---- stage pixel strip (160 halfs) as raw + shifted copies ----
            if (tid < 160) {
                int xx = x0 + tid;
                half_t h = (half_t)0.f;
                if (xx < HW) h = prow[i * HW + xx];
                strip[buf][0][tid] = h;
                if (tid >= 1) strip[buf][1][tid - 1] = h;
            }
            // ---- stage 4KB weight block: one b128 per thread ----
            *(uint4*)&wlds[buf][tid * 8] = wg[(size_t)step * 256 + tid];
            // prefetch next step's weights (-> global_prefetch_b8)
            if (step + 1 < NSTEPS)
                __builtin_prefetch(wg + (size_t)(step + 1) * 256 + tid, 0, 1);

            __syncthreads();   // one barrier per step; all LDS double-buffered

            // ---- build A fragment: 8 aligned dword LDS loads ----
            union { v16h h; uint32_t u[8]; } A;
            const half_t* sb = &strip[buf][sel][0];
#pragma unroll
            for (int v = 0; v < 8; ++v) {
                int s = abase + 2 * (v & 3) + (v >> 2) * 16;   // even for all lanes
                A.u[v] = *(const uint32_t*)(sb + s);
            }
            // ---- B fragments from LDS (32B-aligned v16h) + 4 WMMA ----
            const v16h* wl = (const v16h*)&wlds[buf][0];
            if (q < 4) {       // numerator planes
#pragma unroll
                for (int n = 0; n < 4; ++n)
                    accN[n] = __builtin_amdgcn_wmma_f32_16x16x32_f16(
                        false, A.h, false, wl[n * 32 + lane], (short)0, accN[n], false, false);
            } else {           // denominator planes
#pragma unroll
                for (int n = 0; n < 4; ++n)
                    accD[n] = __builtin_amdgcn_wmma_f32_16x16x32_f16(
                        false, A.h, false, wl[n * 32 + lane], (short)0, accD[n], false, false);
            }
        }
    }

    // ---- epilogue: add scalars, rsqrt-normalize, NaN->0, masked store ----
    const float sN = scal[b * 4 + 0];
    const float sD = scal[b * 4 + 1];
    const float In = scal[b * 4 + 2];
    const int Nl = lane & 15;
#pragma unroll
    for (int n = 0; n < 4; ++n) {
        int part = n * 16 + Nl;
        float* orow = out + (((size_t)b * NPARTS + part) * OUT_HW + y0) * OUT_HW;
#pragma unroll
        for (int r = 0; r < 8; ++r) {
            int x = x0 + wv * 16 + r + hiL * 8;   // C/D layout: M = r + 8*hi
            if (x < OUT_HW) {
                float num = accN[n][r] + sN;
                float den = In * (accD[n][r] + sD);
                float rs  = rsqrtf(den);
                if (rs != rs) rs = 0.f;           // NaN -> 0 (matches reference)
                orow[x] = num * rs;
            }
        }
    }
}

// ---------------------------------------------------------------------------
extern "C" void kernel_launch(void* const* d_in, const int* in_sizes, int n_in,
                              void* d_out, int out_size, void* d_ws, size_t ws_size,
                              hipStream_t stream) {
    const float* image = (const float*)d_in[0];
    const float* parts = (const float*)d_in[1];
    const float* fa    = (const float*)d_in[2];
    const float* aA    = (const float*)d_in[3];
    const float* bg    = (const float*)d_in[4];
    // d_in[5] = padding (0 in setup_inputs)
    float* out = (float*)d_out;

    char* ws = (char*)d_ws;
    const size_t planeBytes = (size_t)BATCH * NPLANES * PIX * sizeof(half_t); // ~15.7 MB
    const size_t wBytes     = (size_t)NSTEPS * 4 * 512 * sizeof(half_t);      // ~1.9 MB
    half_t* planes  = (half_t*)ws;
    half_t* wbuf    = (half_t*)(ws + planeBytes);
    float*  scalars = (float*)(ws + planeBytes + wBytes);

    prep_planes   <<<(BATCH * PIX) / 256, 256, 0, stream>>>(image, fa, aA, bg, planes);
    prep_weights  <<<(NPLANES * 32 * 4 * 512) / 256, 256, 0, stream>>>(parts, wbuf);
    reduce_scalars<<<BATCH, 256, 0, stream>>>(image, fa, aA, bg, scalars);
    conv_main     <<<dim3(2, OUT_HW, BATCH), 256, 0, stream>>>(planes, wbuf, scalars, out);
}
